// EncoderLayer_57578331570209
// MI455X (gfx1250) — matrix operand order repaired and checked by reading, not compiled, not run on hardware
//
#include <hip/hip_runtime.h>
#include <cstdint>
#include <cstddef>

// ---------------------------------------------------------------------------
// Problem constants (match reference)
// ---------------------------------------------------------------------------
#define B_   4
#define S_   2048
#define D_   1024
#define H_   16
#define HD_  64
#define F_   4096
#define E_   8
#define T_   (B_ * S_)
#define EPS_ 1e-5f
#define NEG_ -1e10f

// ---------------------------------------------------------------------------
// CDNA5 WMMA types
// ---------------------------------------------------------------------------
typedef __bf16 bf16_t;
typedef __attribute__((ext_vector_type(16))) __bf16 v16bf;
typedef __attribute__((ext_vector_type(8)))  __bf16 v8bf;
typedef __attribute__((ext_vector_type(2)))  __bf16 v2bf;
typedef __attribute__((ext_vector_type(8)))  float  v8f;
typedef __attribute__((ext_vector_type(4)))  int    v4i_t;

// -------- feature probes (compile-safe) ------------------------------------
#if defined(__has_builtin)
#  if __has_builtin(__builtin_amdgcn_global_load_async_to_lds_b128)
#    define HAVE_ASYNC_LDS 1
#  else
#    define HAVE_ASYNC_LDS 0
#  endif
#  if __has_builtin(__builtin_amdgcn_cvt_pk_bf16_f32)
#    define HAVE_PK_BF16 1
#  else
#    define HAVE_PK_BF16 0
#  endif
#else
#  define HAVE_ASYNC_LDS 0
#  define HAVE_PK_BF16 0
#endif

#if HAVE_ASYNC_LDS
typedef __attribute__((address_space(1))) v4i_t* g_v4i_p;   // global int4*
typedef __attribute__((address_space(3))) v4i_t* l_v4i_p;   // LDS int4* (32-bit)
#endif

__device__ __forceinline__ unsigned short f2bf(float f) {
    unsigned u = __float_as_uint(f);
    u += 0x7FFFu + ((u >> 16) & 1u);     // round-to-nearest-even
    return (unsigned short)(u >> 16);
}

// Pack two f32 -> packed bf16 dword (v_cvt_pk_bf16_f32 when available).
__device__ __forceinline__ unsigned pack_bf16(float a, float b) {
#if HAVE_PK_BF16
    v2bf r = __builtin_amdgcn_cvt_pk_bf16_f32(a, b);
    return __builtin_bit_cast(unsigned, r);
#else
    return (unsigned)f2bf(a) | ((unsigned)f2bf(b) << 16);
#endif
}

// 16-byte global -> LDS copy; async (GLOBAL_LOAD_ASYNC_TO_LDS_B128) if present.
// Note: low 32 bits of a generic pointer to LDS are the LDS byte offset
// (ISA aperture rule), so the integer round-trip is exact.
__device__ __forceinline__ void copy16_g2l(const void* g, void* l) {
#if HAVE_ASYNC_LDS
    __builtin_amdgcn_global_load_async_to_lds_b128(
        (g_v4i_p)(uintptr_t)g,
        (l_v4i_p)(unsigned)(uintptr_t)l, 0, 0);
#else
    *(uint4*)l = *(const uint4*)g;
#endif
}

__device__ __forceinline__ void async_wait0() {
#if HAVE_ASYNC_LDS
#  if __has_builtin(__builtin_amdgcn_s_wait_asynccnt)
    __builtin_amdgcn_s_wait_asynccnt(0);
#  else
    asm volatile("s_wait_asynccnt 0" ::: "memory");
#  endif
#endif
}

__device__ __forceinline__ v16bf cat8(v8bf lo, v8bf hi) {
    return __builtin_shufflevector(lo, hi, 0,1,2,3,4,5,6,7,8,9,10,11,12,13,14,15);
}

__device__ __forceinline__ v8f wmma_bf16(v16bf a, v16bf b, v8f c) {
    return __builtin_amdgcn_wmma_f32_16x16x32_bf16(false, a, false, b,
                                                   (short)0, c, false, false);
}

__device__ __forceinline__ float rmax16(float v) {
    v = fmaxf(v, __shfl_xor(v, 1, 32));
    v = fmaxf(v, __shfl_xor(v, 2, 32));
    v = fmaxf(v, __shfl_xor(v, 4, 32));
    v = fmaxf(v, __shfl_xor(v, 8, 32));
    return v;
}
__device__ __forceinline__ float rsum16(float v) {
    v += __shfl_xor(v, 1, 32);
    v += __shfl_xor(v, 2, 32);
    v += __shfl_xor(v, 4, 32);
    v += __shfl_xor(v, 8, 32);
    return v;
}

// ---------------------------------------------------------------------------
// Tile loaders (f32 -> bf16 conversion on the fly; optional row gather+scale)
// ---------------------------------------------------------------------------
__device__ __forceinline__ void load_tile_A(const float* __restrict__ A,
                                            unsigned short* As,
                                            int m0, int mEnd, int K, int kt, int tid,
                                            const int* __restrict__ rowIdx,
                                            const float* __restrict__ rowScale) {
#pragma unroll
    for (int i = 0; i < 4; ++i) {
        int idx = tid + i * 256;
        int row = idx >> 3;
        int c4  = (idx & 7) * 4;
        int gm  = m0 + row;
        float4 v = make_float4(0.f, 0.f, 0.f, 0.f);
        float  sc = 1.f;
        if (gm < mEnd) {
            long ar = gm;
            if (rowIdx) { ar = rowIdx[gm]; sc = rowScale[ar]; }
            v = *(const float4*)(A + (size_t)ar * K + (size_t)kt * 32 + c4);
        }
        uint2 o;
        o.x = pack_bf16(v.x * sc, v.y * sc);
        o.y = pack_bf16(v.z * sc, v.w * sc);
        *(uint2*)&As[row * 32 + c4] = o;
    }
}

__device__ __forceinline__ void load_tile_A(const unsigned short* __restrict__ A,
                                            unsigned short* As,
                                            int m0, int mEnd, int K, int kt, int tid,
                                            const int* __restrict__,
                                            const float* __restrict__) {
#pragma unroll
    for (int i = 0; i < 2; ++i) {
        int idx = tid + i * 256;
        int row = idx >> 2;
        int c8  = (idx & 3) * 8;
        int gm  = m0 + row;
        if (gm < mEnd) {
            copy16_g2l(A + (size_t)gm * K + (size_t)kt * 32 + c8,
                       &As[row * 32 + c8]);
        } else {
            *(uint4*)&As[row * 32 + c8] = make_uint4(0u, 0u, 0u, 0u);
        }
    }
}

__device__ __forceinline__ void storeC(float* p, float v) { *p = v; }
__device__ __forceinline__ void storeC(unsigned short* p, float v) { *p = f2bf(v); }

// ---------------------------------------------------------------------------
// Generic tiled WMMA GEMM:  C[M,N] = act(A[M,K] * W[N,K]^T + bias)
// Block tile 128x128, K-step 32, 8 waves (4x2), wave tile 32x64.
// Optional expert grouping (offs[e]..offs[e+1]) and row gather+scale.
// ---------------------------------------------------------------------------
template <typename AT, typename OT>
__global__ __launch_bounds__(256)
void gemm_wmma(const AT* __restrict__ A, const float* __restrict__ W,
               const float* __restrict__ bias, OT* __restrict__ C,
               int M, int N, int K,
               const int* __restrict__ offs,
               const int* __restrict__ rowIdx, const float* __restrict__ rowScale,
               long long wstride, long long bstride, int relu) {
    __shared__ __align__(32) unsigned short As[128 * 32];
    __shared__ __align__(32) unsigned short Bs[128 * 32];

    const int e      = blockIdx.z;
    const int mStart = offs ? offs[e]     : 0;
    const int mEnd   = offs ? offs[e + 1] : M;
    const int m0     = mStart + blockIdx.x * 128;
    if (m0 >= mEnd) return;
    const int n0 = blockIdx.y * 128;

    const float* Wp = W    + (size_t)e * (size_t)wstride;
    const float* bp = bias + (size_t)e * (size_t)bstride;

    const int tid  = threadIdx.x;
    const int wid  = tid >> 5;
    const int lane = tid & 31;
    const int wm   = wid & 3;          // 4 wave rows  (32 M each)
    const int wn   = wid >> 2;         // 2 wave cols  (64 N each)
    const int l15  = lane & 15;
    const int hh   = lane >> 4;

    v8f acc[2][4];
#pragma unroll
    for (int tm = 0; tm < 2; ++tm)
#pragma unroll
        for (int tn = 0; tn < 4; ++tn)
#pragma unroll
            for (int r = 0; r < 8; ++r) acc[tm][tn][r] = 0.f;

    const int nk = K >> 5;
    for (int kt = 0; kt < nk; ++kt) {
        load_tile_A(A, As, m0, mEnd, K, kt, tid, rowIdx, rowScale);
#pragma unroll
        for (int i = 0; i < 4; ++i) {
            int idx = tid + i * 256;
            int row = idx >> 3;
            int c4  = (idx & 7) * 4;
            float4 v = *(const float4*)(Wp + (size_t)(n0 + row) * K +
                                        (size_t)kt * 32 + c4);
            uint2 o;
            o.x = pack_bf16(v.x, v.y);
            o.y = pack_bf16(v.z, v.w);
            *(uint2*)&Bs[row * 32 + c4] = o;
        }
        if (kt + 1 < nk)   // speculative prefetch of next weight tile (global_prefetch)
            __builtin_prefetch(Wp + (size_t)(n0 + (tid >> 1)) * K +
                               (size_t)(kt + 1) * 32, 0, 0);
        async_wait0();
        __syncthreads();

#pragma unroll
        for (int tm = 0; tm < 2; ++tm) {
            int arow = wm * 32 + tm * 16 + l15;
            v16bf a  = *(const v16bf*)&As[arow * 32 + hh * 16];
#pragma unroll
            for (int tn = 0; tn < 4; ++tn) {
                int brow = wn * 64 + tn * 16 + l15;
                v16bf b  = *(const v16bf*)&Bs[brow * 32 + hh * 16];
                acc[tm][tn] = wmma_bf16(a, b, acc[tm][tn]);
            }
        }
        __syncthreads();
    }

#pragma unroll
    for (int tm = 0; tm < 2; ++tm)
#pragma unroll
        for (int tn = 0; tn < 4; ++tn) {
            int n   = n0 + wn * 64 + tn * 16 + l15;
            float bv = bp[n];
#pragma unroll
            for (int r = 0; r < 8; ++r) {
                int m = m0 + wm * 32 + tm * 16 + r + hh * 8;
                if (m < mEnd) {
                    float v = acc[tm][tn][r] + bv;
                    if (relu) v = fmaxf(v, 0.f);
                    storeC(&C[(size_t)m * N + n], v);
                }
            }
        }
}

// ---------------------------------------------------------------------------
// Flash attention: block = 64 queries of one (b,h); 4 waves x 16 queries.
// S = QK^T and O += PV both via v_wmma_f32_16x16x32_bf16.
// Q/K tiles are raw bf16 copies -> async global->LDS path.
// ---------------------------------------------------------------------------
__global__ __launch_bounds__(128)
void attn_kernel(const unsigned short* __restrict__ q,
                 const unsigned short* __restrict__ k,
                 const unsigned short* __restrict__ v,
                 const int* __restrict__ mask,
                 unsigned short* __restrict__ ctx) {
    __shared__ __align__(32) unsigned short Qs[64 * 64];
    __shared__ __align__(32) unsigned short Ks[64 * 64];
    __shared__ __align__(32) unsigned short Vt[64 * 64];       // [hd][key]
    __shared__ __align__(32) unsigned short Ps[4 * 16 * 64];   // per-wave P tiles

    const int qb = blockIdx.x, hd = blockIdx.y, bb = blockIdx.z;
    const int tid = threadIdx.x, w = tid >> 5, lane = tid & 31;
    const int l15 = lane & 15, hl = lane >> 4;

#pragma unroll
    for (int i = 0; i < 4; ++i) {                 // load Q tile (async copy)
        int idx = tid + i * 128;
        int row = idx >> 3, c8 = (idx & 7) * 8;
        int s = qb * 64 + row;
        copy16_g2l(q + (((size_t)(bb * S_ + s) * H_ + hd) * HD_ + c8),
                   &Qs[row * 64 + c8]);
    }

    float mrow[8], lrow[8];
    v8f oacc[4];
#pragma unroll
    for (int r = 0; r < 8; ++r) { mrow[r] = -1e30f; lrow[r] = 0.f; }
#pragma unroll
    for (int dt = 0; dt < 4; ++dt)
#pragma unroll
        for (int r = 0; r < 8; ++r) oacc[dt][r] = 0.f;

    for (int kb = 0; kb < S_ / 64; ++kb) {
        __syncthreads();
#pragma unroll
        for (int i = 0; i < 4; ++i) {             // K tile (async) + V^T tile
            int idx = tid + i * 128;
            int row = idx >> 3, c8 = (idx & 7) * 8;
            int s = kb * 64 + row;
            size_t base = ((size_t)(bb * S_ + s) * H_ + hd) * HD_ + c8;
            copy16_g2l(k + base, &Ks[row * 64 + c8]);
            uint4 vv = *(const uint4*)(v + base);
            const unsigned short* ve = (const unsigned short*)&vv;
#pragma unroll
            for (int j = 0; j < 8; ++j) Vt[(c8 + j) * 64 + row] = ve[j];
        }
        async_wait0();
        __syncthreads();

        v8f sacc[4];
#pragma unroll
        for (int nt = 0; nt < 4; ++nt)
#pragma unroll
            for (int r = 0; r < 8; ++r) sacc[nt][r] = 0.f;

#pragma unroll
        for (int ks = 0; ks < 2; ++ks) {          // S = Q K^T   (K-dim = 64)
            int arow = w * 16 + l15;
            v16bf a  = *(const v16bf*)&Qs[arow * 64 + ks * 32 + hl * 16];
#pragma unroll
            for (int nt = 0; nt < 4; ++nt) {
                int brow = nt * 16 + l15;
                v16bf b  = *(const v16bf*)&Ks[brow * 64 + ks * 32 + hl * 16];
                sacc[nt] = wmma_bf16(a, b, sacc[nt]);
            }
        }

        float corr[8];
#pragma unroll
        for (int r = 0; r < 8; ++r) {             // online softmax update
            float rm = -3e38f;
#pragma unroll
            for (int nt = 0; nt < 4; ++nt) {
                int kg = kb * 64 + nt * 16 + l15;
                float sv = sacc[nt][r] * 0.125f;  // 1/sqrt(HD)
                if (mask[(size_t)bb * S_ + kg] == 0) sv = NEG_;
                sacc[nt][r] = sv;
                rm = fmaxf(rm, sv);
            }
            rm = rmax16(rm);
            float mnew = fmaxf(mrow[r], rm);
            float c    = __expf(mrow[r] - mnew);
            float rs   = 0.f;
#pragma unroll
            for (int nt = 0; nt < 4; ++nt) {
                float p = __expf(sacc[nt][r] - mnew);
                rs += p;
                Ps[(w * 16 + r + hl * 8) * 64 + nt * 16 + l15] = f2bf(p);
            }
            rs = rsum16(rs);
            lrow[r] = lrow[r] * c + rs;
            mrow[r] = mnew;
            corr[r] = c;
        }
#pragma unroll
        for (int dt = 0; dt < 4; ++dt)
#pragma unroll
            for (int r = 0; r < 8; ++r) oacc[dt][r] *= corr[r];

        asm volatile("s_wait_dscnt 0" ::: "memory");  // P LDS stores -> A-frag loads

#pragma unroll
        for (int ks = 0; ks < 2; ++ks) {          // O += P V
            int prow = w * 16 + l15;
            v16bf a  = *(const v16bf*)&Ps[prow * 64 + ks * 32 + hl * 16];
#pragma unroll
            for (int dt = 0; dt < 4; ++dt) {
                int brow = dt * 16 + l15;
                v16bf b  = *(const v16bf*)&Vt[brow * 64 + ks * 32 + hl * 16];
                oacc[dt] = wmma_bf16(a, b, oacc[dt]);
            }
        }
    }

#pragma unroll
    for (int dt = 0; dt < 4; ++dt)
#pragma unroll
        for (int r = 0; r < 8; ++r) {
            int m = r + hl * 8;
            int s = qb * 64 + w * 16 + m;
            int d = dt * 16 + l15;
            float ov = oacc[dt][r] / lrow[r];
            ctx[((size_t)(bb * S_ + s) * H_ + hd) * HD_ + d] = f2bf(ov);
        }
}

// ---------------------------------------------------------------------------
// LayerNorm with fused residual: out = LN(a + b) * g + beta
// ---------------------------------------------------------------------------
__global__ __launch_bounds__(256)
void ln_kernel(const float* __restrict__ a, const float* __restrict__ b,
               const float* __restrict__ g, const float* __restrict__ beta,
               float* __restrict__ out) {
    __shared__ float rs[256], rq[256];
    __shared__ float s_mean, s_inv;
    const int t = blockIdx.x, tid = threadIdx.x;
    float v[4], s = 0.f, qq = 0.f;
#pragma unroll
    for (int i = 0; i < 4; ++i) {
        int d = tid + i * 256;
        float x = a[(size_t)t * D_ + d] + b[(size_t)t * D_ + d];
        v[i] = x; s += x; qq += x * x;
    }
    rs[tid] = s; rq[tid] = qq;
    __syncthreads();
    for (int off = 128; off > 0; off >>= 1) {
        if (tid < off) { rs[tid] += rs[tid + off]; rq[tid] += rq[tid + off]; }
        __syncthreads();
    }
    if (tid == 0) {
        float mean = rs[0] / (float)D_;
        float var  = rq[0] / (float)D_ - mean * mean;
        s_mean = mean;
        s_inv  = rsqrtf(var + EPS_);
    }
    __syncthreads();
    float mean = s_mean, inv = s_inv;
#pragma unroll
    for (int i = 0; i < 4; ++i) {
        int d = tid + i * 256;
        out[(size_t)t * D_ + d] = (v[i] - mean) * inv * g[d] + beta[d];
    }
}

// ---------------------------------------------------------------------------
// Router: logits = x1 @ switch_w^T + switch_b; pmax = max softmax prob;
// routes = argmax (first occurrence on ties, like jnp.argmax).
// ---------------------------------------------------------------------------
__global__ __launch_bounds__(64)
void router_kernel(const float* __restrict__ x1, const float* __restrict__ sw,
                   const float* __restrict__ sb, int* __restrict__ routes,
                   float* __restrict__ pmax) {
    __shared__ float red[64 * E_];
    const int t = blockIdx.x, tid = threadIdx.x;
    float acc[E_] = {};
    for (int d = tid; d < D_; d += 64) {
        float xv = x1[(size_t)t * D_ + d];
#pragma unroll
        for (int e = 0; e < E_; ++e) acc[e] += xv * sw[e * D_ + d];
    }
#pragma unroll
    for (int e = 0; e < E_; ++e) red[tid * E_ + e] = acc[e];
    __syncthreads();
    if (tid == 0) {
        float lg[E_];
#pragma unroll
        for (int e = 0; e < E_; ++e) {
            float s = 0.f;
            for (int i = 0; i < 64; ++i) s += red[i * E_ + e];
            lg[e] = s + sb[e];
        }
        float mx = lg[0]; int bi = 0;
#pragma unroll
        for (int e = 1; e < E_; ++e)
            if (lg[e] > mx) { mx = lg[e]; bi = e; }
        float den = 0.f;
#pragma unroll
        for (int e = 0; e < E_; ++e) den += __expf(lg[e] - mx);
        routes[t] = bi;
        pmax[t]   = 1.f / den;
    }
}

// ---------------------------------------------------------------------------
// Stable counting sort by expert id (matches argsort(routes, stable=True)).
// ---------------------------------------------------------------------------
__global__ void sort_kernel(const int* __restrict__ routes,
                            int* __restrict__ order, int* __restrict__ offs) {
    if (threadIdx.x != 0 || blockIdx.x != 0) return;
    int cnt[E_];
    for (int e = 0; e < E_; ++e) cnt[e] = 0;
    for (int t = 0; t < T_; ++t) cnt[routes[t]]++;
    int off[E_ + 1];
    off[0] = 0;
    for (int e = 0; e < E_; ++e) off[e + 1] = off[e] + cnt[e];
    for (int e = 0; e <= E_; ++e) offs[e] = off[e];
    int cur[E_];
    for (int e = 0; e < E_; ++e) cur[e] = off[e];
    for (int t = 0; t < T_; ++t) order[cur[routes[t]]++] = t;
}

// ---------------------------------------------------------------------------
// Launch
// ---------------------------------------------------------------------------
extern "C" void kernel_launch(void* const* d_in, const int* in_sizes, int n_in,
                              void* d_out, int out_size, void* d_ws, size_t ws_size,
                              hipStream_t stream) {
    (void)in_sizes; (void)n_in; (void)out_size; (void)ws_size;
    const float* x    = (const float*)d_in[0];
    const float* wq   = (const float*)d_in[1];
    const float* bq   = (const float*)d_in[2];
    const float* wk   = (const float*)d_in[3];
    const float* bk   = (const float*)d_in[4];
    const float* wv   = (const float*)d_in[5];
    const float* bv   = (const float*)d_in[6];
    const float* wo   = (const float*)d_in[7];
    const float* bo   = (const float*)d_in[8];
    const float* ln1g = (const float*)d_in[9];
    const float* ln1b = (const float*)d_in[10];
    const float* ln2g = (const float*)d_in[11];
    const float* ln2b = (const float*)d_in[12];
    const float* sw   = (const float*)d_in[13];
    const float* sb   = (const float*)d_in[14];
    const float* ew1  = (const float*)d_in[15];
    const float* eb1  = (const float*)d_in[16];
    const float* ew2  = (const float*)d_in[17];
    const float* eb2  = (const float*)d_in[18];
    const int*   mask = (const int*)d_in[19];
    float* out = (float*)d_out;

    char* wsp = (char*)d_ws;
    auto alloc = [&](size_t bytes) -> char* {
        char* p = wsp;
        wsp += (bytes + 255) & ~(size_t)255;
        return p;
    };
    unsigned short* qb   = (unsigned short*)alloc((size_t)T_ * D_ * 2);
    unsigned short* kb   = (unsigned short*)alloc((size_t)T_ * D_ * 2);
    unsigned short* vb   = (unsigned short*)alloc((size_t)T_ * D_ * 2);
    unsigned short* ctxb = (unsigned short*)alloc((size_t)T_ * D_ * 2);
    float*          aout = (float*)alloc((size_t)T_ * D_ * 4);
    float*          x1   = (float*)alloc((size_t)T_ * D_ * 4);
    unsigned short* hbuf = (unsigned short*)alloc((size_t)T_ * F_ * 2);
    float*          moe  = (float*)alloc((size_t)T_ * D_ * 4);
    int*            routes = (int*)alloc((size_t)T_ * 4);
    float*          pmaxv  = (float*)alloc((size_t)T_ * 4);
    int*            order  = (int*)alloc((size_t)T_ * 4);
    int*            offs   = (int*)alloc((E_ + 1) * 4);

    const dim3 gProj(T_ / 128, D_ / 128, 1);

    // --- QKV projections (f32 -> bf16 WMMA, bf16 out) ---
    gemm_wmma<float, unsigned short><<<gProj, 256, 0, stream>>>(
        x, wq, bq, qb, T_, D_, D_, nullptr, nullptr, nullptr, 0, 0, 0);
    gemm_wmma<float, unsigned short><<<gProj, 256, 0, stream>>>(
        x, wk, bk, kb, T_, D_, D_, nullptr, nullptr, nullptr, 0, 0, 0);
    gemm_wmma<float, unsigned short><<<gProj, 256, 0, stream>>>(
        x, wv, bv, vb, T_, D_, D_, nullptr, nullptr, nullptr, 0, 0, 0);

    // --- flash attention ---
    attn_kernel<<<dim3(S_ / 64, H_, B_), 128, 0, stream>>>(qb, kb, vb, mask, ctxb);

    // --- output projection ---
    gemm_wmma<unsigned short, float><<<gProj, 256, 0, stream>>>(
        ctxb, wo, bo, aout, T_, D_, D_, nullptr, nullptr, nullptr, 0, 0, 0);

    // --- LN1 (x + attn_out) ---
    ln_kernel<<<T_, 256, 0, stream>>>(x, aout, ln1g, ln1b, x1);

    // --- router + stable expert sort ---
    router_kernel<<<T_, 64, 0, stream>>>(x1, sw, sb, routes, pmaxv);
    sort_kernel<<<1, 32, 0, stream>>>(routes, order, offs);

    // --- MoE grouped GEMM1: h = relu(xs @ W1^T + b1), gather+pmax fused ---
    gemm_wmma<float, unsigned short><<<dim3(T_ / 128, F_ / 128, E_), 256, 0, stream>>>(
        x1, ew1, eb1, hbuf, T_, F_, D_, offs, order, pmaxv,
        (long long)F_ * D_, (long long)F_, 1);

    // --- MoE grouped GEMM2: moe = h @ W2^T + b2 (written in permuted position) ---
    gemm_wmma<unsigned short, float><<<dim3(T_ / 128, D_ / 128, E_), 256, 0, stream>>>(
        hbuf, ew2, eb2, moe, T_, D_, F_, offs, nullptr, nullptr,
        (long long)D_ * F_, (long long)D_, 0);

    // --- LN2 (x1 + moe) -> out ---
    ln_kernel<<<T_, 256, 0, stream>>>(x1, moe, ln2g, ln2b, out);
}
